// SimpleRNN_32641751449606
// MI455X (gfx1250) — matrix operand order repaired and checked by reading, not compiled, not run on hardware
//
#include <hip/hip_runtime.h>

// ---------------------------------------------------------------------------
// Fused Dense + GRU + Linear head for (B=2048, S=512, F_IN=15, H=32).
//
//   Wc = W_ih @ W_dense  (96x15) built per block; biases folded into K=15
//   column of Wc with x[k=15] == 1.0  (r/z tiles also absorb b_hh there).
//   Per step, per wave (16 batch rows):  12x v_wmma_f32_16x16x32_f16:
//     j=0..3 (r,z):  acc = wmma(x, Wc_j, 0) ; acc = wmma(h, Whh_j, acc)
//     j=4,5  (n):    gx  = wmma(x, Wc_j, 0) ; gh  = wmma(h, Whh_j, b_hh_j)
//   h/x A-fragments staged in wave-private LDS in *fragment order* so each
//   lane's 16 halves are one contiguous 32B load (no register shuffling).
// ---------------------------------------------------------------------------

typedef _Float16 v16h __attribute__((ext_vector_type(16)));
typedef float    v8f  __attribute__((ext_vector_type(8)));

#define SEQ   512
#define FIN   15
#define HID   32
#define GATES 96
#define TAIL  24
#define WPB   8          // waves per block
#define RPW   16         // batch rows per wave (WMMA M)

__device__ __forceinline__ float fast_sigmoid(float x) {
  return __builtin_amdgcn_rcpf(1.0f + __builtin_amdgcn_exp2f(-1.44269504f * x));
}
__device__ __forceinline__ float fast_tanh(float x) {
  // tanh(x) = 2*sigmoid(2x) - 1
  return 2.0f * __builtin_amdgcn_rcpf(1.0f + __builtin_amdgcn_exp2f(-2.88539008f * x)) - 1.0f;
}

// ISA 7.12.2 16-bit A layout, (m,k) -> flat index lane*16+i:
//   lane = m + 16*((k>>3)&1) ; i = (k&7) + 8*((k>>4)&1)
__device__ __forceinline__ int fragAddr(int m, int k) {
  return (m + ((k & 8) ? 16 : 0)) * 16 + (k & 7) + ((k & 16) ? 8 : 0);
}

// A-fragment from frag-ordered staging buffer: one contiguous 32B load.
__device__ __forceinline__ v16h load_fragA(const _Float16* buf, int lane) {
  return *(const v16h*)(buf + lane * 16);
}

// B-fragment (K=32 x N=16) from N-major matrix (stride 32): lane holds column
// n = L%16, halves K = 0..7,16..23 (lanes 0-15) / 8..15,24..31 (lanes 16-31).
__device__ __forceinline__ v16h load_fragB(const _Float16* baseT, int lane) {
  const int n  = lane & 15;
  const int k0 = (lane & 16) ? 8 : 0;
  const _Float16* p = baseT + n * 32 + k0;
  v16h r;
#pragma unroll
  for (int i = 0; i < 8; ++i) { r[i] = p[i]; r[i + 8] = p[i + 16]; }
  return r;
}

__device__ __forceinline__ v8f vfill(float v) {
  v8f o;
#pragma unroll
  for (int i = 0; i < 8; ++i) o[i] = v;
  return o;
}

__global__ __launch_bounds__(WPB * 32, 1)
void gru_fused_kernel(const float* __restrict__ x,        // (B,S,15)
                      const float* __restrict__ W_dense,  // (32,15)
                      const float* __restrict__ b_dense,  // (32)
                      const float* __restrict__ W_ih,     // (96,32)
                      const float* __restrict__ W_hh,     // (96,32)
                      const float* __restrict__ b_ih,     // (96)
                      const float* __restrict__ b_hh,     // (96)
                      const float* __restrict__ W_out,    // (1,32)
                      const float* __restrict__ b_out,    // (1)
                      float* __restrict__ out)            // (B,24,1)
{
  __shared__ __align__(32) _Float16 s_whh[GATES * 32];     // W_hh [n][k]
  __shared__ __align__(32) _Float16 s_wc [GATES * 32];     // Wc   [n][k] + bias col 15
  __shared__ __align__(32) _Float16 s_x  [WPB][RPW * 32];  // frag-ordered x tile
  __shared__ __align__(32) _Float16 s_h  [WPB][RPW * 32];  // frag-ordered h tile
  __shared__ float s_bhh[GATES], s_wout[32], s_bout;

  const int t = threadIdx.x;

  // ---- per-block weight prep (redundant across blocks, trivial cost) ----
  for (int i = t; i < GATES * 32; i += WPB * 32) {
    s_whh[i] = (_Float16)W_hh[i];
    const int n = i >> 5, k = i & 31;
    float acc = 0.0f;
    if (k < FIN) {
#pragma unroll
      for (int h = 0; h < HID; ++h) acc += W_ih[n * HID + h] * W_dense[h * FIN + k];
    } else if (k == FIN) {
      // bias column, multiplied by the pinned x[k=15] == 1.0
      acc = b_ih[n];
#pragma unroll
      for (int h = 0; h < HID; ++h) acc += W_ih[n * HID + h] * b_dense[h];
      if (n < 64) acc += b_hh[n];          // r,z gates absorb b_hh here too
    }
    s_wc[i] = (_Float16)acc;
  }
  for (int i = t; i < GATES; i += WPB * 32) s_bhh[i] = b_hh[i];
  if (t < 32) s_wout[t] = W_out[t];
  if (t == 0) s_bout = b_out[0];
  // zero staging buffers (x padding cols stay 0; h starts at 0)
  for (int i = t; i < WPB * RPW * 32; i += WPB * 32) {
    (&s_x[0][0])[i] = (_Float16)0.0f;
    (&s_h[0][0])[i] = (_Float16)0.0f;
  }
  __syncthreads();

  const int lane = t & 31;
  const int wave = t >> 5;
  const int b0   = (blockIdx.x * WPB + wave) * RPW;
  _Float16* xl = &s_x[wave][0];
  _Float16* hl = &s_h[wave][0];

  // pin x[k=15] = 1.0 for every row (bias slot; never overwritten below)
  if (lane < RPW) xl[fragAddr(lane, FIN)] = (_Float16)1.0f;
  __builtin_amdgcn_wave_barrier();

  // ---- resident weight fragments ----
  v16h whhf[6], wcf[6];
#pragma unroll
  for (int j = 0; j < 6; ++j) {
    whhf[j] = load_fragB(s_whh + j * 16 * 32, lane);
    wcf[j]  = load_fragB(s_wc  + j * 16 * 32, lane);
  }
  const v8f bh4 = vfill(s_bhh[64 + (lane & 15)]);
  const v8f bh5 = vfill(s_bhh[80 + (lane & 15)]);
  const v8f vzero = vfill(0.0f);

  // ---- per-lane x-tile gather plan: 240 f32, branchless (clamped dup) ----
  // 32-bit byte offsets (x spans only 63 MB), advanced by 60B per step so the
  // loads lower to saddr-base + u32-voffset form with one v_add each.
  unsigned off[8];
  int      lbo[8];
#pragma unroll
  for (int i = 0; i < 8; ++i) {
    int e = i * 32 + lane;
    if (e >= RPW * FIN) e = RPW * FIN - 1;          // duplicate elem 239: benign
    const int row = e / FIN, col = e - row * FIN;
    off[i] = (unsigned)(((b0 + row) * (SEQ * FIN) + col) * 4);
    lbo[i] = fragAddr(row, col);
  }
  const char* xb = (const char*)x;

  float xr[8];
#pragma unroll
  for (int i = 0; i < 8; ++i) xr[i] = *(const float*)(xb + off[i]);

  // per-lane h store base (D-layout (m = r+mlo, n) -> frag order)
  const int mlo = (lane & 16) ? 8 : 0;
  const int nn  = lane & 15;
  const int hsb = (mlo + ((nn & 8) ? 16 : 0)) * 16 + (nn & 7);

  v8f h0 = vfill(0.0f), h1 = vfill(0.0f);   // hidden cols 0-15 / 16-31

  for (int s = 0; s < SEQ; ++s) {
    // stage x tile into fragment order
#pragma unroll
    for (int i = 0; i < 8; ++i) xl[lbo[i]] = (_Float16)xr[i];
    __builtin_amdgcn_wave_barrier();

    // prefetch next step's x while this step computes (uniform guard keeps the
    // last iteration from walking past the end of x; re-reads step 511)
    if (s < SEQ - 1) {
#pragma unroll
      for (int i = 0; i < 8; ++i) off[i] += FIN * 4;
    }
#pragma unroll
    for (int i = 0; i < 8; ++i) xr[i] = *(const float*)(xb + off[i]);

    const v16h xf = load_fragA(xl, lane);
    const v16h hf = load_fragA(hl, lane);

    // r,z tiles: chained accumulation, biases already inside Wc col 15
    v8f a0 = __builtin_amdgcn_wmma_f32_16x16x32_f16(false, xf, false, wcf[0], (short)0, vzero, false, false);
    v8f a1 = __builtin_amdgcn_wmma_f32_16x16x32_f16(false, xf, false, wcf[1], (short)0, vzero, false, false);
    v8f a2 = __builtin_amdgcn_wmma_f32_16x16x32_f16(false, xf, false, wcf[2], (short)0, vzero, false, false);
    v8f a3 = __builtin_amdgcn_wmma_f32_16x16x32_f16(false, xf, false, wcf[3], (short)0, vzero, false, false);
    a0 = __builtin_amdgcn_wmma_f32_16x16x32_f16(false, hf, false, whhf[0], (short)0, a0, false, false);
    a1 = __builtin_amdgcn_wmma_f32_16x16x32_f16(false, hf, false, whhf[1], (short)0, a1, false, false);
    a2 = __builtin_amdgcn_wmma_f32_16x16x32_f16(false, hf, false, whhf[2], (short)0, a2, false, false);
    a3 = __builtin_amdgcn_wmma_f32_16x16x32_f16(false, hf, false, whhf[3], (short)0, a3, false, false);
    // n tiles: xn and hn kept separate (r gates hn before tanh)
    v8f gx4 = __builtin_amdgcn_wmma_f32_16x16x32_f16(false, xf, false, wcf[4], (short)0, vzero, false, false);
    v8f gx5 = __builtin_amdgcn_wmma_f32_16x16x32_f16(false, xf, false, wcf[5], (short)0, vzero, false, false);
    v8f gh4 = __builtin_amdgcn_wmma_f32_16x16x32_f16(false, hf, false, whhf[4], (short)0, bh4, false, false);
    v8f gh5 = __builtin_amdgcn_wmma_f32_16x16x32_f16(false, hf, false, whhf[5], (short)0, bh5, false, false);

#pragma unroll
    for (int i = 0; i < 8; ++i) {
      const float r0 = fast_sigmoid(a0[i]);
      const float r1 = fast_sigmoid(a1[i]);
      const float z0 = fast_sigmoid(a2[i]);
      const float z1 = fast_sigmoid(a3[i]);
      const float n0 = fast_tanh(gx4[i] + r0 * gh4[i]);
      const float n1 = fast_tanh(gx5[i] + r1 * gh5[i]);
      h0[i] = n0 + z0 * (h0[i] - n0);     // (1-z)*n + z*h, fma form
      h1[i] = n1 + z1 * (h1[i] - n1);
    }

    // stage h for next step's A-fragment (D-layout -> frag order)
#pragma unroll
    for (int r = 0; r < 8; ++r) {
      hl[hsb + r * 16]     = (_Float16)h0[r];
      hl[hsb + r * 16 + 8] = (_Float16)h1[r];
    }
    __builtin_amdgcn_wave_barrier();

    // output head for the last 24 steps
    if (s >= SEQ - TAIL) {
      if (lane < RPW) {
        float acc = s_bout;
#pragma unroll
        for (int k = 0; k < HID; ++k)
          acc += (float)hl[fragAddr(lane, k)] * s_wout[k];
        out[(long)(b0 + lane) * TAIL + (s - (SEQ - TAIL))] = acc;
      }
      __builtin_amdgcn_wave_barrier();
    }
  }
}

extern "C" void kernel_launch(void* const* d_in, const int* in_sizes, int n_in,
                              void* d_out, int out_size, void* d_ws, size_t ws_size,
                              hipStream_t stream) {
  (void)in_sizes; (void)n_in; (void)d_ws; (void)ws_size; (void)out_size;
  const float* x       = (const float*)d_in[0];
  const float* W_dense = (const float*)d_in[1];
  const float* b_dense = (const float*)d_in[2];
  const float* W_ih    = (const float*)d_in[3];
  const float* W_hh    = (const float*)d_in[4];
  const float* b_ih    = (const float*)d_in[5];
  const float* b_hh    = (const float*)d_in[6];
  const float* W_out   = (const float*)d_in[7];
  const float* b_out   = (const float*)d_in[8];
  float* out           = (float*)d_out;

  const int B = 2048;
  const int rows_per_block = WPB * RPW;            // 128
  dim3 grid(B / rows_per_block);                   // 16 blocks
  dim3 block(WPB * 32);                            // 8 wave32s
  gru_fused_kernel<<<grid, block, 0, stream>>>(x, W_dense, b_dense, W_ih, W_hh,
                                               b_ih, b_hh, W_out, b_out, out);
}
